// MyLSTMold_73478300500607
// MI455X (gfx1250) — matrix operand order repaired and checked by reading, not compiled, not run on hardware
//
#include <hip/hip_runtime.h>
#include <hip/hip_bf16.h>
#include <math.h>

// ---------------------------------------------------------------------------
// Problem constants (from reference)
// ---------------------------------------------------------------------------
#define B_      2048
#define T_      50
#define H_      256
#define CMD_    2
#define LIDAR_  360
#define IN_     362            // CMD + LIDAR
#define KX_     384            // IN_ padded to multiple of 32 (WMMA K)
#define NOUTP_  368            // LIDAR padded to multiple of 16 (WMMA N)
#define PFJ_    24             // 384/16 floats staged per thread per step

typedef __attribute__((ext_vector_type(16))) __bf16 bf16x16;
typedef __attribute__((ext_vector_type(8)))  __bf16 bf16x8;
typedef __attribute__((ext_vector_type(8)))  float  f32x8;

__device__ __forceinline__ float sigf(float x) { return 1.0f / (1.0f + __expf(-x)); }

__device__ __forceinline__ bf16x16 cat16(bf16x8 lo, bf16x8 hi) {
    return __builtin_shufflevector(lo, hi, 0,1,2,3,4,5,6,7,8,9,10,11,12,13,14,15);
}

// A-tile (16x32, MxK) for v_wmma_*_16x16x32_bf16 from row-major bf16 [16][stride].
// 16-bit A layout: lane L -> M = L%16; half = L/16.
//   elements 0..7  hold K = k0 + half*8 + e
//   elements 8..15 hold K = k0 + 16 + half*8 + (e-8)
// -> two contiguous 16B chunks per lane (ds_load_b128 x2 from LDS).
__device__ __forceinline__ bf16x16 load_a_lds(const __bf16* base, int stride, int lane, int k0) {
    int r = lane & 15, half = lane >> 4;
    const __bf16* p = base + r * stride + k0 + half * 8;
    bf16x8 lo = *(const bf16x8*)(p);
    bf16x8 hi = *(const bf16x8*)(p + 16);
    return cat16(lo, hi);
}

// B-tile (32x16, KxN): B[k][n] = W[n][k] with W packed row-major [N][Kpad] bf16.
// Lane L: N = n0 + L%16, elements 0..7 hold K = k0 + (L/16)*8 + e, elements
// 8..15 hold K = k0 + 16 + (L/16)*8 + (e-8) (global_load_b128 x2, L2-resident weights).
__device__ __forceinline__ bf16x16 load_b_w(const __bf16* W, int Kpad, int lane, int n0, int k0) {
    int n = n0 + (lane & 15), half = lane >> 4;
    const __bf16* p = W + (size_t)n * Kpad + k0 + half * 8;
    bf16x8 lo = *(const bf16x8*)(p);
    bf16x8 hi = *(const bf16x8*)(p + 16);
    return cat16(lo, hi);
}

__device__ __forceinline__ f32x8 wmma_bf16(bf16x16 a, bf16x16 b, f32x8 c) {
    // D = A*B + C ; f32 accumulate
    return __builtin_amdgcn_wmma_f32_16x16x32_bf16(false, a, false, b, (short)0, c, false, false);
}

// ---------------------------------------------------------------------------
// Weight pre-pack: fp32 [N][Ks] -> bf16 [Npad][Kpad], zero padded
// ---------------------------------------------------------------------------
__global__ void pack_w_bf16(const float* __restrict__ src, unsigned short* __restrict__ dst,
                            int N, int Ks, int Npad, int Kpad) {
    int i = blockIdx.x * blockDim.x + threadIdx.x;
    int total = Npad * Kpad;
    if (i >= total) return;
    int n = i / Kpad, k = i % Kpad;
    float v = (n < N && k < Ks) ? src[(size_t)n * Ks + k] : 0.0f;
    __bf16 h = (__bf16)v;
    dst[i] = __builtin_bit_cast(unsigned short, h);
}

// ---------------------------------------------------------------------------
// Persistent seq2seq LSTM kernel.
// One workgroup (256 threads = 8 waves, wave32) owns a 16-row batch tile for
// the full 50 encode + 50 decode steps. Wave w owns hidden columns
// [w*32, w*32+32): 4 gates x 2 WMMA N-tiles. c stays in registers in C-layout,
// h is shared via bf16 LDS (A-operand of the Whh GEMM). Encoder inputs are
// double-buffered: x(t+1) is register-prefetched at the top of step t so the
// global-load latency overlaps the 160 WMMAs of the current step.
// ---------------------------------------------------------------------------
__global__ __launch_bounds__(256)
void lstm_seq2seq(const float* __restrict__ past, const float* __restrict__ fcmd,
                  const float* __restrict__ enc_b, const float* __restrict__ dec_b,
                  const float* __restrict__ out_b,
                  const unsigned short* encWih_u, const unsigned short* encWhh_u,
                  const unsigned short* decWih_u, const unsigned short* decWhh_u,
                  const unsigned short* outW_u, float* __restrict__ out)
{
    const __bf16* encWih = (const __bf16*)encWih_u;
    const __bf16* encWhh = (const __bf16*)encWhh_u;
    const __bf16* decWih = (const __bf16*)decWih_u;
    const __bf16* decWhh = (const __bf16*)decWhh_u;
    const __bf16* outW   = (const __bf16*)outW_u;

    __shared__ __bf16 xbf[2][16][KX_];  // double-buffered input tile, bf16
    __shared__ __bf16 hbf[16][H_];      // hidden state, bf16 (WMMA A source)
    __shared__ float  lid[16][NOUTP_];  // decoder lidar feedback, fp32

    const int tid  = threadIdx.x;
    const int lane = tid & 31;
    const int w    = tid >> 5;          // wave id 0..7
    const int b0   = blockIdx.x * 16;   // batch-tile origin
    const int hc0  = w * 32;            // hidden columns owned by this wave
    const int mrow = (lane >> 4) * 8;   // C-layout: VGPR r holds row mrow+r
    const int ncol = lane & 15;         // C-layout: column within tile

    // staging map: 16 threads per row, 24 contiguous floats per thread
    const int srow = tid >> 4;
    const int skb  = (tid & 15) * PFJ_;

    // warm L2 with the first weight matrix (global_prefetch_b8)
    __builtin_prefetch(encWih_u + (size_t)tid * 64, 0, 0);

    for (int i = tid; i < 16 * H_; i += 256)
        hbf[i / H_][i % H_] = (__bf16)0.0f;

    f32x8 cst[2];
    #pragma unroll
    for (int j = 0; j < 2; ++j)
        #pragma unroll
        for (int r = 0; r < 8; ++r) cst[j][r] = 0.0f;

    // prologue: stage x(0) into xbf[0]
    {
        const float* src = &past[((size_t)(b0 + srow) * T_ + 0) * IN_];
        #pragma unroll
        for (int j = 0; j < PFJ_; ++j) {
            int k = skb + j;
            float v = (k < IN_) ? src[k] : 0.0f;
            xbf[0][srow][k] = (__bf16)v;
        }
    }
    __syncthreads();

    // ------------------------------- encoder -------------------------------
    for (int t = 0; t < T_; ++t) {
        const int cur = t & 1;

        // register-prefetch x(t+1); loads overlap the WMMA phase below
        float pf[PFJ_];
        const bool havepf = (t + 1 < T_);
        if (havepf) {
            const float* src = &past[((size_t)(b0 + srow) * T_ + (t + 1)) * IN_];
            #pragma unroll
            for (int j = 0; j < PFJ_; ++j) {
                int k = skb + j;
                pf[j] = (k < IN_) ? src[k] : 0.0f;
            }
        }

        f32x8 acc[4][2];
        #pragma unroll
        for (int g = 0; g < 4; ++g)
            #pragma unroll
            for (int j = 0; j < 2; ++j) {
                float bb = enc_b[g * H_ + hc0 + j * 16 + ncol];
                #pragma unroll
                for (int r = 0; r < 8; ++r) acc[g][j][r] = bb;
            }

        for (int kt = 0; kt < KX_; kt += 32) {
            bf16x16 a = load_a_lds(&xbf[cur][0][0], KX_, lane, kt);
            #pragma unroll
            for (int g = 0; g < 4; ++g)
                #pragma unroll
                for (int j = 0; j < 2; ++j)
                    acc[g][j] = wmma_bf16(a, load_b_w(encWih, KX_, lane,
                                          g * H_ + hc0 + j * 16, kt), acc[g][j]);
        }
        for (int kt = 0; kt < H_; kt += 32) {
            bf16x16 a = load_a_lds(&hbf[0][0], H_, lane, kt);
            #pragma unroll
            for (int g = 0; g < 4; ++g)
                #pragma unroll
                for (int j = 0; j < 2; ++j)
                    acc[g][j] = wmma_bf16(a, load_b_w(encWhh, H_, lane,
                                          g * H_ + hc0 + j * 16, kt), acc[g][j]);
        }
        __syncthreads();   // all waves done reading xbf[cur] / hbf

        #pragma unroll
        for (int j = 0; j < 2; ++j)
            #pragma unroll
            for (int r = 0; r < 8; ++r) {
                float cn = sigf(acc[1][j][r]) * cst[j][r]
                         + sigf(acc[0][j][r]) * tanhf(acc[2][j][r]);
                cst[j][r] = cn;
                float hn = sigf(acc[3][j][r]) * tanhf(cn);
                hbf[mrow + r][hc0 + j * 16 + ncol] = (__bf16)hn;
            }

        // drain prefetch into the other buffer (nobody reads it this step)
        if (havepf) {
            #pragma unroll
            for (int j = 0; j < PFJ_; ++j)
                xbf[1 - cur][srow][skb + j] = (__bf16)pf[j];
        }
        __syncthreads();   // new h + x(t+1) visible
    }

    // ----------------------------- decoder init ----------------------------
    for (int i = tid; i < 16 * NOUTP_; i += 256) {
        int r = i / NOUTP_, k = i % NOUTP_;
        lid[r][k] = (k < LIDAR_)
                  ? past[((size_t)(b0 + r) * T_ + (T_ - 1)) * IN_ + CMD_ + k]
                  : 0.0f;
    }
    __syncthreads();

    // ------------------------------- decoder -------------------------------
    for (int t = 0; t < T_; ++t) {
        // build x = [lidar_feedback, cmd] in bf16 (lid is LDS, cmd is tiny)
        {
            #pragma unroll
            for (int j = 0; j < PFJ_; ++j) {
                int k = skb + j;
                float v;
                if (k < LIDAR_)   v = lid[srow][k];
                else if (k < IN_) v = fcmd[((size_t)(b0 + srow) * T_ + t) * CMD_ + (k - LIDAR_)];
                else              v = 0.0f;
                xbf[0][srow][k] = (__bf16)v;
            }
        }
        __syncthreads();

        f32x8 acc[4][2];
        #pragma unroll
        for (int g = 0; g < 4; ++g)
            #pragma unroll
            for (int j = 0; j < 2; ++j) {
                float bb = dec_b[g * H_ + hc0 + j * 16 + ncol];
                #pragma unroll
                for (int r = 0; r < 8; ++r) acc[g][j][r] = bb;
            }

        for (int kt = 0; kt < KX_; kt += 32) {
            bf16x16 a = load_a_lds(&xbf[0][0][0], KX_, lane, kt);
            #pragma unroll
            for (int g = 0; g < 4; ++g)
                #pragma unroll
                for (int j = 0; j < 2; ++j)
                    acc[g][j] = wmma_bf16(a, load_b_w(decWih, KX_, lane,
                                          g * H_ + hc0 + j * 16, kt), acc[g][j]);
        }
        for (int kt = 0; kt < H_; kt += 32) {
            bf16x16 a = load_a_lds(&hbf[0][0], H_, lane, kt);
            #pragma unroll
            for (int g = 0; g < 4; ++g)
                #pragma unroll
                for (int j = 0; j < 2; ++j)
                    acc[g][j] = wmma_bf16(a, load_b_w(decWhh, H_, lane,
                                          g * H_ + hc0 + j * 16, kt), acc[g][j]);
        }
        __syncthreads();   // done reading xbf / hbf

        #pragma unroll
        for (int j = 0; j < 2; ++j)
            #pragma unroll
            for (int r = 0; r < 8; ++r) {
                float cn = sigf(acc[1][j][r]) * cst[j][r]
                         + sigf(acc[0][j][r]) * tanhf(acc[2][j][r]);
                cst[j][r] = cn;
                float hn = sigf(acc[3][j][r]) * tanhf(cn);
                hbf[mrow + r][hc0 + j * 16 + ncol] = (__bf16)hn;
            }
        __syncthreads();   // new h visible to all waves

        // output projection: lidar = h @ out_W^T + out_b (16 x 360, 23 N-tiles)
        #pragma unroll
        for (int jj = 0; jj < 3; ++jj) {
            int tile = w + jj * 8;
            if (tile * 16 >= NOUTP_) break;    // tiles 0..22
            int n0 = tile * 16;
            int n  = n0 + ncol;
            f32x8 po;
            float bb = (n < LIDAR_) ? out_b[n] : 0.0f;
            #pragma unroll
            for (int r = 0; r < 8; ++r) po[r] = bb;
            for (int kt = 0; kt < H_; kt += 32) {
                bf16x16 a = load_a_lds(&hbf[0][0], H_, lane, kt);
                po = wmma_bf16(a, load_b_w(outW, H_, lane, n0, kt), po);
            }
            #pragma unroll
            for (int r = 0; r < 8; ++r) {
                lid[mrow + r][n] = po[r];      // feedback for next step
                if (n < LIDAR_)
                    out[((size_t)(b0 + mrow + r) * T_ + t) * LIDAR_ + n] = po[r];
            }
        }
        __syncthreads();   // lid complete before next step's x build
    }
}

// ---------------------------------------------------------------------------
// Launch
// ---------------------------------------------------------------------------
extern "C" void kernel_launch(void* const* d_in, const int* in_sizes, int n_in,
                              void* d_out, int out_size, void* d_ws, size_t ws_size,
                              hipStream_t stream) {
    const float* past   = (const float*)d_in[0];
    const float* fcmd   = (const float*)d_in[1];
    const float* encWih = (const float*)d_in[2];
    const float* encWhh = (const float*)d_in[3];
    const float* enc_b  = (const float*)d_in[4];
    const float* decWih = (const float*)d_in[5];
    const float* decWhh = (const float*)d_in[6];
    const float* dec_b  = (const float*)d_in[7];
    const float* outW   = (const float*)d_in[8];
    const float* out_b  = (const float*)d_in[9];

    // bf16 packed weights in workspace (~2.8 MB total, L2 resident)
    unsigned short* ws        = (unsigned short*)d_ws;
    unsigned short* encWih_pk = ws;                          // 1024*384
    unsigned short* encWhh_pk = encWih_pk + 1024 * 384;      // 1024*256
    unsigned short* decWih_pk = encWhh_pk + 1024 * 256;      // 1024*384
    unsigned short* decWhh_pk = decWih_pk + 1024 * 384;      // 1024*256
    unsigned short* outW_pk   = decWhh_pk + 1024 * 256;      // 368*256

    struct Job { const float* s; unsigned short* d; int N, Ks, Np, Kp; };
    Job jobs[5] = {
        { encWih, encWih_pk, 1024, IN_, 1024, KX_ },
        { encWhh, encWhh_pk, 1024, H_,  1024, H_  },
        { decWih, decWih_pk, 1024, IN_, 1024, KX_ },
        { decWhh, decWhh_pk, 1024, H_,  1024, H_  },
        { outW,   outW_pk,   LIDAR_, H_, NOUTP_, H_ },
    };
    for (int i = 0; i < 5; ++i) {
        int tot = jobs[i].Np * jobs[i].Kp;
        pack_w_bf16<<<(tot + 255) / 256, 256, 0, stream>>>(
            jobs[i].s, jobs[i].d, jobs[i].N, jobs[i].Ks, jobs[i].Np, jobs[i].Kp);
    }

    lstm_seq2seq<<<B_ / 16, 256, 0, stream>>>(
        past, fcmd, enc_b, dec_b, out_b,
        encWih_pk, encWhh_pk, decWih_pk, decWhh_pk, outW_pk,
        (float*)d_out);
}
